// WeightedPermuteMLP_83442624626908
// MI455X (gfx1250) — compile-verified, matrix operand order checked by reading, not run on hardware
//
#include <hip/hip_runtime.h>
#include <hip/hip_bf16.h>

// CDNA5 / gfx1250: wave32, WMMA 16x16x32 bf16 with fp32 accumulation.
typedef __attribute__((ext_vector_type(16))) __bf16 v16bf;
typedef __attribute__((ext_vector_type(8)))  float  v8f;

// x: (8,16,16,16,1024)  flat strides: b:4194304 h:262144 w:16384 d:1024 c:1
// tokens NTOK=32768, C=1024, seg=16, S=64

__device__ __forceinline__ unsigned int f2bf(float f) {
  unsigned int u = __float_as_uint(f);
  return (u + 0x7FFFu + ((u >> 16) & 1u)) >> 16;   // round-to-nearest-even
}

// ---------------- pass 0: conversions / packing ----------------

__global__ void k_cvt_x(const float* __restrict__ x, unsigned int* __restrict__ xb, int nPairs) {
  int i = blockIdx.x * blockDim.x + threadIdx.x;
  if (i < nPairs) {
    float2 v = ((const float2*)x)[i];
    xb[i] = f2bf(v.x) | (f2bf(v.y) << 16);
  }
}

// Pack 1024x1024 weights into WMMA B-fragment order:
// tile ti = kt*64+nt (K-tile of 32, N-tile of 16); per lane 8 contiguous u32:
// u32 j = pair (K = kt*32 + (lane&16) + 2j, 2j+1) at column n = nt*16 + (lane&15)
__global__ void k_pack_w(const float* __restrict__ wh, const float* __restrict__ ww,
                         const float* __restrict__ wd, const float* __restrict__ wc,
                         const float* __restrict__ wp, unsigned int* __restrict__ pw) {
  const int wi = blockIdx.y;
  const float* W = (wi == 0) ? wh : (wi == 1) ? ww : (wi == 2) ? wd : (wi == 3) ? wc : wp;
  const int p = blockIdx.x * blockDim.x + threadIdx.x;  // < 524288
  if (p >= 524288) return;
  const int j = p & 7, lane = (p >> 3) & 31, ti = p >> 8;
  const int nt = ti & 63, kt = ti >> 6;
  const int k0 = kt * 32 + (lane & 16) + 2 * j;
  const int nn = nt * 16 + (lane & 15);
  pw[wi * 524288 + p] = f2bf(W[k0 * 1024 + nn]) | (f2bf(W[(k0 + 1) * 1024 + nn]) << 16);
}

// ---------------- pass 1: attention statistics (cheap scalar path) ----------------

__global__ void k_xbar(const float* __restrict__ x, float* __restrict__ xWD,
                       float* __restrict__ xHD, float* __restrict__ xHW) {
  const int id = blockIdx.x * blockDim.x + threadIdx.x;  // 131072
  if (id >= 131072) return;
  const int cc = id & 1023, i = (id >> 10) & 15, b = id >> 14;
  const float* xp = x + (long)b * 4194304;
  float sWD = 0.f, sHD = 0.f, sHW = 0.f;
  for (int p = 0; p < 256; ++p) {
    const int a = p >> 4, q = p & 15;
    sWD += xp[(long)i * 262144 + a * 16384 + q * 1024 + cc];
    sHD += xp[(long)a * 262144 + i * 16384 + q * 1024 + cc];
    sHW += xp[(long)a * 262144 + q * 16384 + i * 1024 + cc];
  }
  xWD[id] = sWD * (1.f / 256.f);
  xHD[id] = sHD * (1.f / 256.f);
  xHW[id] = sHW * (1.f / 256.f);
}

__global__ void k_xall(const float* __restrict__ xWD, float* __restrict__ xAll) {
  const int id = blockIdx.x * blockDim.x + threadIdx.x;  // 8192
  if (id >= 8192) return;
  const int k = id & 1023, b = id >> 10;
  float a = 0.f;
  for (int i = 0; i < 16; ++i) a += xWD[b * 16384 + i * 1024 + k];
  xAll[id] = a * (1.f / 16.f);
}

__global__ void k_wr(const float* __restrict__ wh, const float* __restrict__ ww,
                     const float* __restrict__ wd, float* __restrict__ wr) {
  const int id = blockIdx.x * blockDim.x + threadIdx.x;  // 196608
  if (id >= 196608) return;
  const int s = id & 63, K = (id >> 6) & 1023, wsel = id >> 16;
  const float* W = (wsel == 0) ? wh : (wsel == 1) ? ww : wd;
  float a = 0.f;
  for (int hh = 0; hh < 16; ++hh) a += W[K * 1024 + hh * 64 + s];
  wr[id] = a * (1.f / 16.f);
}

__global__ void k_abar(const float* __restrict__ xWD, const float* __restrict__ xHD,
                       const float* __restrict__ xHW, const float* __restrict__ xAll,
                       const float* __restrict__ wr, const float* __restrict__ wc,
                       const float* __restrict__ bh, const float* __restrict__ bw,
                       const float* __restrict__ bd, const float* __restrict__ bc,
                       float* __restrict__ abar) {
  const int id = blockIdx.x * blockDim.x + threadIdx.x;  // 8192
  if (id >= 8192) return;
  const int c = id & 1023, b = id >> 10;
  const int si = c >> 6, s = c & 63;
  const float* whr = wr;
  const float* wwr = wr + 65536;
  const float* wdr = wr + 131072;
  float acc = 0.f;
  for (int K = 0; K < 1024; ++K) {
    const int hp = K >> 6, sp = K & 63;
    const int xi = b * 16384 + hp * 1024 + si * 64 + sp;
    acc += xWD[xi] * whr[K * 64 + s]
         + xHD[xi] * wwr[K * 64 + s]
         + xHW[xi] * wdr[K * 64 + s]
         + xAll[(b << 10) + K] * wc[K * 1024 + c];
  }
  float bsum = 0.f;
  for (int hh = 0; hh < 16; ++hh) bsum += bh[hh * 64 + s] + bw[hh * 64 + s] + bd[hh * 64 + s];
  abar[id] = acc + bsum * (1.f / 16.f) + bc[c];
}

__global__ void k_mlp1(const float* __restrict__ abar, const float* __restrict__ w1,
                       const float* __restrict__ b1, float* __restrict__ tb) {
  const int id = blockIdx.x * blockDim.x + threadIdx.x;  // 2048
  if (id >= 2048) return;
  const int j = id & 255, b = id >> 8;
  float a = b1[j];
  for (int k = 0; k < 1024; ++k) a += abar[(b << 10) + k] * w1[k * 256 + j];
  tb[id] = 0.5f * a * (1.f + erff(a * 0.7071067811865476f));  // exact GELU
}

__global__ void k_attn(const float* __restrict__ tb, const float* __restrict__ w2,
                       const float* __restrict__ b2, float4* __restrict__ attn4) {
  const int id = blockIdx.x * blockDim.x + threadIdx.x;  // 8192
  if (id >= 8192) return;
  const int c = id & 1023, b = id >> 10;
  float z0 = b2[c * 4 + 0], z1 = b2[c * 4 + 1], z2 = b2[c * 4 + 2], z3 = b2[c * 4 + 3];
  for (int j = 0; j < 256; ++j) {
    const float t = tb[(b << 8) + j];
    const float* wrow = w2 + (long)j * 4096 + c * 4;
    z0 += t * wrow[0]; z1 += t * wrow[1]; z2 += t * wrow[2]; z3 += t * wrow[3];
  }
  const float mx = fmaxf(fmaxf(z0, z1), fmaxf(z2, z3));
  const float e0 = expf(z0 - mx), e1 = expf(z1 - mx), e2 = expf(z2 - mx), e3 = expf(z3 - mx);
  const float inv = 1.f / (e0 + e1 + e2 + e3);
  attn4[id] = make_float4(e0 * inv, e1 * inv, e2 * inv, e3 * inv);  // (h,w,d,c)
}

// ---------------- pass 2: fused 4-branch WMMA GEMMs + attention combine ----------------

// One 16x16x1024 tile-GEMM: 32 K-steps, A gathered from xb, B from packed weights.
__device__ __forceinline__ v8f tile_gemm(const unsigned short* __restrict__ xb,
                                         const unsigned short* __restrict__ W,
                                         long aBase, int strideK, int nt,
                                         int lane, int half) {
  v8f acc = {0.f, 0.f, 0.f, 0.f, 0.f, 0.f, 0.f, 0.f};
#pragma unroll 4
  for (int kt = 0; kt < 32; ++kt) {
    const int K0 = kt * 32 + half;
    const v16bf Af = *(const v16bf*)(xb + aBase + (long)(K0 >> 6) * strideK + (K0 & 63));
    const v16bf Bf = *(const v16bf*)(W + (((long)(kt * 64 + nt) * 32 + lane) << 4));
    acc = __builtin_amdgcn_wmma_f32_16x16x32_bf16(false, Af, false, Bf, (short)0, acc, false, false);
  }
  return acc;
}

// region = (b, H0, W0, s-stripe s0) = 16 tokens (d) x 256 channels; one wave per region.
__global__ void __launch_bounds__(64) k_regions(
    const unsigned short* __restrict__ xb, const unsigned short* __restrict__ pw,
    const float* __restrict__ attnf,   // attn as flat floats: [(b*1024+c)*4 + comp]
    const float* __restrict__ bh, const float* __restrict__ bw,
    const float* __restrict__ bd, const float* __restrict__ bc,
    unsigned int* __restrict__ gbuf) {
  extern __shared__ float smem[];
  const int wave = threadIdx.x >> 5;
  const int lane = threadIdx.x & 31;
  const int region = blockIdx.x * 2 + wave;       // 8192 regions
  const int s016 = region & 3, s0 = s016 << 4;
  const int W0 = (region >> 2) & 15;
  const int H0 = (region >> 6) & 15;
  const int b  = region >> 10;
  float* reg = smem + wave * 4096;                // private 16 KB region buffer

  // seed with attention-weighted biases: reg[d*256 + u*16 + j], c = u*64+s0+j
  for (int i = lane; i < 4096; i += 32) {
    const int d = i >> 8, cho = i & 255, u = cho >> 4, j = cho & 15;
    const int c = u * 64 + s0 + j, s = s0 + j;
    const float* av = attnf + (((b << 10) + c) << 2);
    reg[i] = av[0] * bh[H0 * 64 + s] + av[1] * bw[W0 * 64 + s]
           + av[2] * bd[d * 64 + s] + av[3] * bc[c];
  }

  const int m    = lane & 15;
  const int half = lane & 16;
  const int hi8  = (lane >> 4) << 3;
  const int n    = lane & 15;
  const long xbB = (long)b * 4194304;
  const long attnB = (long)(b << 10);

  // ---- branch 0: h-mix (attn comp 0) ----
  {
    const long aBase0 = xbB + (long)W0 * 16384 + (long)m * 1024;
    const int ntFix = H0 * 4 + s016;
    for (int u = 0; u < 16; ++u) {
      const v8f acc = tile_gemm(xb, pw, aBase0 + u * 64, 262144, ntFix, lane, half);
      const float aw = attnf[((attnB + u * 64 + s0 + n) << 2) + 0];
#pragma unroll
      for (int v = 0; v < 8; ++v)
        atomicAdd(&reg[(v + hi8) * 256 + u * 16 + n], aw * acc[v]);
    }
  }
  // ---- branch 1: w-mix (attn comp 1) ----
  {
    const long aBase0 = xbB + (long)H0 * 262144 + (long)m * 1024;
    const int ntFix = W0 * 4 + s016;
    const unsigned short* W = pw + 1048576;
    for (int u = 0; u < 16; ++u) {
      const v8f acc = tile_gemm(xb, W, aBase0 + u * 64, 16384, ntFix, lane, half);
      const float aw = attnf[((attnB + u * 64 + s0 + n) << 2) + 1];
#pragma unroll
      for (int v = 0; v < 8; ++v)
        atomicAdd(&reg[(v + hi8) * 256 + u * 16 + n], aw * acc[v]);
    }
  }
  // ---- branch 2: d-mix (attn comp 2); rows are s_idx, tiles are dd ----
  {
    const long aBase0 = xbB + (long)H0 * 262144 + (long)W0 * 16384 + (long)m * 64;
    const unsigned short* W = pw + 2 * 1048576;
    float awz[8];  // invariant across dd tiles: attn.z at (s_idx = v+hi8, n)
#pragma unroll
    for (int v = 0; v < 8; ++v)
      awz[v] = attnf[((attnB + (v + hi8) * 64 + s0 + n) << 2) + 2];
    for (int u = 0; u < 16; ++u) {
      const v8f acc = tile_gemm(xb, W, aBase0, 1024, u * 4 + s016, lane, half);
#pragma unroll
      for (int v = 0; v < 8; ++v)
        atomicAdd(&reg[u * 256 + (v + hi8) * 16 + n], awz[v] * acc[v]);
    }
  }
  // ---- branch 3: c-mix (attn comp 3) ----
  {
    const long aBase0 = xbB + (long)H0 * 262144 + (long)W0 * 16384 + (long)m * 1024;
    const unsigned short* W = pw + 3 * 1048576;
    for (int u = 0; u < 16; ++u) {
      const v8f acc = tile_gemm(xb, W, aBase0, 64, u * 4 + s016, lane, half);
      const float aw = attnf[((attnB + u * 64 + s0 + n) << 2) + 3];
#pragma unroll
      for (int v = 0; v < 8; ++v)
        atomicAdd(&reg[(v + hi8) * 256 + u * 16 + n], aw * acc[v]);
    }
  }

  // store combined g region as bf16 (row-major token x channel)
  const long tBase = (((long)b * 16 + H0) * 16 + W0) * 16;
  for (int i = lane; i < 2048; i += 32) {
    const int d = i >> 7, r = i & 127, u = r >> 3, jp = r & 7;
    const float f0 = reg[d * 256 + u * 16 + 2 * jp];
    const float f1 = reg[d * 256 + u * 16 + 2 * jp + 1];
    const int c = u * 64 + s0 + 2 * jp;
    gbuf[((((tBase + d) << 10) + c) >> 1)] = f2bf(f0) | (f2bf(f1) << 16);
  }
}

// ---------------- pass 3: out = g @ wp + bp (WMMA GEMM) ----------------
__global__ void __launch_bounds__(128) k_final(
    const unsigned short* __restrict__ gb, const unsigned short* __restrict__ pwp,
    const float* __restrict__ bp, float* __restrict__ out) {
  const int wave = threadIdx.x >> 5;
  const int lane = threadIdx.x & 31;
  const int mt = blockIdx.x >> 2;         // 2048 row tiles of 16
  const int cb = blockIdx.x & 3;
  const int colBase = cb * 256 + wave * 64;
  const int m = lane & 15, half = lane & 16, hi8 = (lane >> 4) << 3, n = lane & 15;
  const long aBase = (((long)mt * 16 + m) << 10);
  const int ntb = colBase >> 4;
  const v8f z = {0.f, 0.f, 0.f, 0.f, 0.f, 0.f, 0.f, 0.f};
  v8f acc[4] = {z, z, z, z};
#pragma unroll 2
  for (int kt = 0; kt < 32; ++kt) {
    const v16bf Af = *(const v16bf*)(gb + aBase + kt * 32 + half);   // reused by 4 tiles
#pragma unroll
    for (int ct = 0; ct < 4; ++ct) {
      const v16bf Bf = *(const v16bf*)(pwp + (((long)(kt * 64 + ntb + ct) * 32 + lane) << 4));
      acc[ct] = __builtin_amdgcn_wmma_f32_16x16x32_bf16(false, Af, false, Bf, (short)0, acc[ct], false, false);
    }
  }
#pragma unroll
  for (int ct = 0; ct < 4; ++ct) {
    const int col = colBase + ct * 16 + n;
    const float bias = bp[col];
#pragma unroll
    for (int v = 0; v < 8; ++v) {
      out[(((long)mt * 16 + v + hi8) << 10) + col] = acc[ct][v] + bias;
    }
  }
}

// ---------------- host ----------------
extern "C" void kernel_launch(void* const* d_in, const int* in_sizes, int n_in,
                              void* d_out, int out_size, void* d_ws, size_t ws_size,
                              hipStream_t stream) {
  (void)in_sizes; (void)n_in; (void)out_size; (void)ws_size;
  const float* x  = (const float*)d_in[0];
  const float* wh = (const float*)d_in[1];
  const float* bh = (const float*)d_in[2];
  const float* ww = (const float*)d_in[3];
  const float* bw = (const float*)d_in[4];
  const float* wd = (const float*)d_in[5];
  const float* bd = (const float*)d_in[6];
  const float* wc = (const float*)d_in[7];
  const float* bc = (const float*)d_in[8];
  const float* w1 = (const float*)d_in[9];
  const float* b1 = (const float*)d_in[10];
  const float* w2 = (const float*)d_in[11];
  const float* b2 = (const float*)d_in[12];
  const float* wp = (const float*)d_in[13];
  const float* bp = (const float*)d_in[14];
  float* out = (float*)d_out;

  char* ws = (char*)d_ws;
  unsigned short* xb = (unsigned short*)ws;                   //  67,108,864 B
  unsigned short* pw = (unsigned short*)(ws + 67108864);      //  10,485,760 B (5 packed weights)
  unsigned short* gb = (unsigned short*)(ws + 77594624);      //  67,108,864 B
  float* xWD  = (float*)(ws + 144703488);                     //  131072 f
  float* xHD  = xWD + 131072;
  float* xHW  = xHD + 131072;
  float* xAll = xHW + 131072;                                 //  8192 f
  float* wr   = xAll + 8192;                                  //  196608 f
  float* abar = wr + 196608;                                  //  8192 f
  float* tb   = abar + 8192;                                  //  2048 f
  float4* attn4 = (float4*)(tb + 2048);                       //  8192 float4

  k_cvt_x<<<65536, 256, 0, stream>>>(x, (unsigned int*)xb, 16777216);
  dim3 gpw(2048, 5);
  k_pack_w<<<gpw, 256, 0, stream>>>(wh, ww, wd, wc, wp, (unsigned int*)pw);
  k_xbar<<<512, 256, 0, stream>>>(x, xWD, xHD, xHW);
  k_xall<<<32, 256, 0, stream>>>(xWD, xAll);
  k_wr<<<768, 256, 0, stream>>>(wh, ww, wd, wr);
  k_abar<<<32, 256, 0, stream>>>(xWD, xHD, xHW, xAll, wr, wc, bh, bw, bd, bc, abar);
  k_mlp1<<<8, 256, 0, stream>>>(abar, w1, b1, tb);
  k_attn<<<32, 256, 0, stream>>>(tb, w2, b2, attn4);
  k_regions<<<4096, 64, 32768, stream>>>(xb, pw, (const float*)attn4, bh, bw, bd, bc,
                                         (unsigned int*)gb);
  k_final<<<8192, 128, 0, stream>>>(gb, pw + 4 * 1048576, bp, out);
}
